// KVLoRA_74534862455241
// MI455X (gfx1250) — matrix operand order repaired and checked by reading, not compiled, not run on hardware
//
#include <hip/hip_runtime.h>

// ---------------------------------------------------------------------------
// KV-LoRA fused QKV projection for MI455X (gfx1250, wave32).
//
// Roofline: 206 GFLOP vs 304 MB compulsory traffic -> need ~15.8 PFLOPS to be
// memory bound. fp32 WMMA (16x16x4) is ~8x slower per instruction than bf16
// (16x16x32), so we convert x and the LoRA-fused weight to bf16 once (both
// then live in the 192MB L2) and run a bf16 WMMA GEMM with f32 accumulation.
// Tile fills use the CDNA5 Tensor Data Mover (tensor_load_to_lds) with LDS
// double buffering: DMA of K-step k+1 overlaps the 8 WMMAs of K-step k.
//
// Workspace layout (needs >= 56 MB):
//   [0, 32MB)   x as bf16    [8192 x 2048]
//   [32, 56MB)  W+Delta bf16 [6144 x 2048]
// ---------------------------------------------------------------------------

typedef __bf16  bf16_t;
typedef bf16_t  v16bf __attribute__((ext_vector_type(16)));
typedef bf16_t  v8bf  __attribute__((ext_vector_type(8)));
typedef float   v8f   __attribute__((ext_vector_type(8)));
typedef unsigned int v4u __attribute__((ext_vector_type(4)));
typedef int          v4i __attribute__((ext_vector_type(4)));
typedef int          v8i __attribute__((ext_vector_type(8)));

static constexpr int Md = 8192;   // B*S rows of x
static constexpr int Kd = 2048;   // in_features
static constexpr int Nd = 6144;   // 3*d fused output rows
static constexpr int Dd = 2048;   // per-head dim
static constexpr int Rr = 8;      // LoRA rank
static constexpr int NK = Kd / 32;

#if __has_builtin(__builtin_amdgcn_tensor_load_to_lds) && \
    __has_builtin(__builtin_amdgcn_s_wait_tensorcnt)
#define USE_TDM 1
#else
#define USE_TDM 0
#endif

__device__ __forceinline__ unsigned int f2bf_rne(float f) {
  unsigned int u = __float_as_uint(f);
  u += 0x7fffu + ((u >> 16) & 1u);   // round-to-nearest-even
  return u >> 16;
}

// ---------------- prep: x fp32 -> bf16 (8 elements / thread) ----------------
__global__ __launch_bounds__(256) void kvlora_prep_x(
    const float* __restrict__ x, unsigned short* __restrict__ xbf) {
  const int gid = blockIdx.x * 256 + threadIdx.x;
  const float4* src = (const float4*)x;
  float4 a = src[2 * gid + 0];
  float4 c = src[2 * gid + 1];
  uint4 o;
  o.x = f2bf_rne(a.x) | (f2bf_rne(a.y) << 16);
  o.y = f2bf_rne(a.z) | (f2bf_rne(a.w) << 16);
  o.z = f2bf_rne(c.x) | (f2bf_rne(c.y) << 16);
  o.w = f2bf_rne(c.z) | (f2bf_rne(c.w) << 16);
  ((uint4*)xbf)[gid] = o;
}

// ------- prep: Wf = bf16(W + [0; B1@A1; B2@A2]) (4 elements / thread) -------
__global__ __launch_bounds__(256) void kvlora_prep_w(
    const float* __restrict__ W,
    const float* __restrict__ A1, const float* __restrict__ B1,
    const float* __restrict__ A2, const float* __restrict__ B2,
    unsigned short* __restrict__ wbf) {
  const int gid = blockIdx.x * 256 + threadIdx.x;   // one per 4 consecutive k
  const int o  = gid >> 9;          // Kd/4 = 512 quads per row
  const int kq = gid & 511;
  float4 w = ((const float4*)W)[gid];
  if (o >= Dd) {                     // K/V rows get the rank-8 delta
    const float* Arow;
    const float* Brow;
    if (o < 2 * Dd) { Brow = B1 + (size_t)(o - Dd) * Rr;     Arow = A1; }
    else            { Brow = B2 + (size_t)(o - 2 * Dd) * Rr; Arow = A2; }
    const int k = kq * 4;
#pragma unroll
    for (int j = 0; j < Rr; ++j) {
      const float bj = Brow[j];
      const float* Aj = Arow + (size_t)j * Kd + k;
      w.x += bj * Aj[0];
      w.y += bj * Aj[1];
      w.z += bj * Aj[2];
      w.w += bj * Aj[3];
    }
  }
  uint2 p;
  p.x = f2bf_rne(w.x) | (f2bf_rne(w.y) << 16);
  p.y = f2bf_rne(w.z) | (f2bf_rne(w.w) << 16);
  ((uint2*)wbf)[gid] = p;
}

#if USE_TDM
// Issue one TDM 2D tile load: tile_d0=32 bf16 along K (64B rows, padded to
// 80B in LDS via pad_interval=64B/pad_amount=16B), tile_d1=128 rows.
// D# bit layout per CDNA5 ISA 8.3-8.5 (group0 128b, group1 256b).
__device__ __forceinline__ void tdm_load_tile(
    unsigned lds_off, const unsigned short* gsrc,
    unsigned tensor_d1 /* rows in tensor */) {
  unsigned long long ga = (unsigned long long)(size_t)gsrc;
  v4u g0;
  g0.x = 0x1u;                                  // count=1, user descriptor
  g0.y = lds_off;                               // lds_addr (bytes)
  g0.z = (unsigned)(ga & 0xffffffffu);          // global_addr[31:0]
  g0.w = (unsigned)((ga >> 32) & 0x01ffffffu)   // global_addr[56:32]
         | (2u << 30);                          // type = 2 ("image")
  v8i g1;
  g1[0] = (int)((1u << 16)     // data_size = 1 -> 2-byte elements
              | (1u << 20)     // pad_enable
              | (3u << 22)     // pad_interval code 3 -> every 16 DW (64B)
              | (3u << 25));   // pad_amount  code 3 -> 4 DW (16B)
  g1[1] = (int)((unsigned)Kd << 16);                    // tensor_dim0 lo16
  g1[2] = (int)(((unsigned)Kd >> 16) | (tensor_d1 << 16)); // d0 hi | d1 lo
  g1[3] = (int)((tensor_d1 >> 16) | (32u << 16));       // d1 hi | tile_dim0=32
  g1[4] = (int)(128u);                                  // tile_dim1=128, tile_dim2=0
  g1[5] = (int)((unsigned)Kd);                          // tensor_dim0_stride lo32
  g1[6] = 0;                                            // stride hi | dim1_stride lo
  g1[7] = 0;
  v4i z4 = {0, 0, 0, 0};
#if defined(__clang_major__) && (__clang_major__ >= 23)
  v8i z8 = {0, 0, 0, 0, 0, 0, 0, 0};
  __builtin_amdgcn_tensor_load_to_lds(g0, g1, z4, z4, z8, 0);
#else
  __builtin_amdgcn_tensor_load_to_lds(g0, g1, z4, z4, 0);
#endif
}
#endif

// ---------------------------- main bf16 WMMA GEMM ---------------------------
// Block tile 128(M) x 128(N), K-step 32. 256 threads = 8 wave32s in a 4x2
// (M x N) wave grid; each wave computes 32x64 = 2x4 tiles of 16x16.
__global__ __launch_bounds__(256) void kvlora_gemm(
    const unsigned short* __restrict__ xbf,
    const unsigned short* __restrict__ wbf,
    const float* __restrict__ bias,
    float* __restrict__ out) {
  // +8 bf16 row pad -> 80B stride: b128 fragment reads are bank-conflict-free
  __shared__ __align__(16) unsigned short lds_a[2][128][40];
  __shared__ __align__(16) unsigned short lds_b[2][128][40];

  const int n0   = blockIdx.x * 128;
  const int m0   = blockIdx.y * 128;
  const int t    = threadIdx.x;
  const int lane = t & 31;
  const int wid  = t >> 5;
  const int wm   = wid >> 1;      // 0..3  (M direction)
  const int wn   = wid & 1;       // 0..1  (N direction)
  const int lr   = lane & 15;     // row/col within 16x16 tile
  const int lh   = lane >> 4;     // lane-half selects K sub-range

  v8f acc[2][4];
#pragma unroll
  for (int i = 0; i < 2; ++i)
#pragma unroll
    for (int j = 0; j < 4; ++j)
      acc[i][j] = (v8f){0.f, 0.f, 0.f, 0.f, 0.f, 0.f, 0.f, 0.f};

  const unsigned short* gA = xbf + (size_t)m0 * Kd;
  const unsigned short* gB = wbf + (size_t)n0 * Kd;

#if USE_TDM
  // Wave 0 drives the Tensor Data Mover; LDS offsets = flat addr low 32 bits.
  const unsigned offA0 = (unsigned)(size_t)&lds_a[0][0][0];
  const unsigned offA1 = (unsigned)(size_t)&lds_a[1][0][0];
  const unsigned offB0 = (unsigned)(size_t)&lds_b[0][0][0];
  const unsigned offB1 = (unsigned)(size_t)&lds_b[1][0][0];

  if (wid == 0) {                       // preload K-step 0 into buffer 0
    tdm_load_tile(offA0, gA, (unsigned)Md);
    tdm_load_tile(offB0, gB, (unsigned)Nd);
  }

  for (int ki = 0; ki < NK; ++ki) {
    const int cur = ki & 1;
    if (wid == 0) {
      if (ki + 1 < NK) {                // kick DMA for next K-step, other buffer
        const int koff = (ki + 1) * 32;
        tdm_load_tile(cur ? offA0 : offA1, gA + koff, (unsigned)Md);
        tdm_load_tile(cur ? offB0 : offB1, gB + koff, (unsigned)Nd);
        __builtin_amdgcn_s_wait_tensorcnt(2);   // retire current pair only
      } else {
        __builtin_amdgcn_s_wait_tensorcnt(0);
      }
    }
    __syncthreads();                    // publish buffer `cur` to all waves
#else
  // Fallback: synchronous cooperative copy (threads 0-127: A, 128-255: B).
  const bool isB = t >= 128;
  const int  row = isB ? (t - 128) : t;
  const unsigned short* gsrc = (isB ? gB : gA) + (size_t)row * Kd;
  for (int ki = 0; ki < NK; ++ki) {
    const int cur = ki & 1;
    const uint4* s = (const uint4*)(gsrc + ki * 32);
    uint4* ldst = isB ? (uint4*)&lds_b[cur][row][0] : (uint4*)&lds_a[cur][row][0];
    ldst[0] = s[0];
    ldst[1] = s[1];
    ldst[2] = s[2];
    ldst[3] = s[3];
    if (ki + 1 < NK)
      __builtin_prefetch(gsrc + (ki + 1) * 32, 0, 3);
    __syncthreads();
#endif

    // A fragments: 16x32 bf16, lane lr = M, VGPRs 0-3 hold K = lh*8..+7,
    // VGPRs 4-7 hold K = lh*8+16..+23 (ISA 16-bit A layout).
    v16bf af[2];
#pragma unroll
    for (int am = 0; am < 2; ++am) {
      const int r = wm * 32 + am * 16 + lr;
      v8bf lo = *(const v8bf*)&lds_a[cur][r][lh * 8];
      v8bf hi = *(const v8bf*)&lds_a[cur][r][lh * 8 + 16];
      af[am] = __builtin_shufflevector(lo, hi, 0, 1, 2, 3, 4, 5, 6, 7,
                                       8, 9, 10, 11, 12, 13, 14, 15);
    }
    // B fragments: 32x16 bf16, lane lr = N, K = lh*8..+7 and lh*8+16..+23
    // (same K order as A), reading a row of Wf (k-contiguous).
    v16bf bfr[4];
#pragma unroll
    for (int bn = 0; bn < 4; ++bn) {
      const int r = wn * 64 + bn * 16 + lr;
      v8bf lo = *(const v8bf*)&lds_b[cur][r][lh * 8];
      v8bf hi = *(const v8bf*)&lds_b[cur][r][lh * 8 + 16];
      bfr[bn] = __builtin_shufflevector(lo, hi, 0, 1, 2, 3, 4, 5, 6, 7,
                                        8, 9, 10, 11, 12, 13, 14, 15);
    }

#pragma unroll
    for (int am = 0; am < 2; ++am)
#pragma unroll
      for (int bn = 0; bn < 4; ++bn)
        acc[am][bn] = __builtin_amdgcn_wmma_f32_16x16x32_bf16(
            /*neg_a=*/false, af[am], /*neg_b=*/false, bfr[bn],
            /*c_mod=*/(short)0, acc[am][bn],
            /*reuse_a=*/false, /*reuse_b=*/false);

    __syncthreads();                    // all reads of `cur` done before reuse
  }

  // Epilogue: C/D layout -> element i of v8f is M = i + 8*lh, N = lr.
#pragma unroll
  for (int am = 0; am < 2; ++am) {
    const int gmb = m0 + wm * 32 + am * 16 + lh * 8;
#pragma unroll
    for (int bn = 0; bn < 4; ++bn) {
      const int gn = n0 + wn * 64 + bn * 16 + lr;
      const float bv = bias[gn];
      v8f c = acc[am][bn];
#pragma unroll
      for (int i = 0; i < 8; ++i)
        out[(size_t)(gmb + i) * Nd + gn] = c[i] + bv;
    }
  }
}

// ---------------------------------------------------------------------------
extern "C" void kernel_launch(void* const* d_in, const int* in_sizes, int n_in,
                              void* d_out, int out_size, void* d_ws, size_t ws_size,
                              hipStream_t stream) {
  const float* x  = (const float*)d_in[0];
  const float* W  = (const float*)d_in[1];
  const float* b  = (const float*)d_in[2];
  const float* A1 = (const float*)d_in[3];
  const float* B1 = (const float*)d_in[4];
  const float* A2 = (const float*)d_in[5];
  const float* B2 = (const float*)d_in[6];
  float* out = (float*)d_out;

  unsigned short* xbf = (unsigned short*)d_ws;            // 32 MB
  unsigned short* wbf = xbf + (size_t)Md * Kd;            // 24 MB

  kvlora_prep_x<<<(Md * Kd) / (8 * 256), 256, 0, stream>>>(x, xbf);
  kvlora_prep_w<<<(Nd * Kd) / (4 * 256), 256, 0, stream>>>(W, A1, B1, A2, B2, wbf);
  kvlora_gemm<<<dim3(Nd / 128, Md / 128), 256, 0, stream>>>(xbf, wbf, b, out);
}
